// SPEncod_80513456931142
// MI455X (gfx1250) — matrix operand order repaired and checked by reading, not compiled, not run on hardware
//
#include <hip/hip_runtime.h>
#include <hip/hip_bf16.h>
#include <math.h>

typedef __bf16 bf16;
typedef __attribute__((ext_vector_type(16))) __bf16 v16bf;
typedef __attribute__((ext_vector_type(8)))  __bf16 v8bf;
typedef __attribute__((ext_vector_type(8)))  float  v8f;

static __device__ __forceinline__ float b2f(bf16 x) { return (float)x; }
static __device__ __forceinline__ bf16  f2b(float x) { return (bf16)x; }

union TileU { v16bf v; v8bf h[2]; unsigned int w[8]; };

static __device__ __forceinline__ v8f wmma_bf16(v16bf a, v16bf b, v8f c) {
  return __builtin_amdgcn_wmma_f32_16x16x32_bf16(false, a, false, b, (short)0, c, false, false);
}

// ---------------------------------------------------------------------------
// Branch-free WMMA tile GEMM.  C(16x16,f32) = A[arow-tile rows, K] * Bt^T.
// CDNA5 16x16x32 bf16 fragment layouts (CDNA5 ISA):
//  A: lane<16 row=lane, elems0..7=K[k0..k0+7], elems8..15=K[k0+16..k0+23];
//     lane>=16: +8 on both windows.
//  B: lane row = output column; same K mapping as A (elems0..7=K[k0+8h..],
//     elems8..15=K[k0+16+8h..], h = lane>=16).
// KPAD = loop bound (multiple of 32, B data must be valid/padded to KPAD);
// KA   = valid A columns (16 allowed: upper chunk compile-time zeroed).
// ---------------------------------------------------------------------------
template<int KPAD, int KA>
static __device__ __forceinline__ v8f wave_gemm_tile(
    const bf16* __restrict__ A, int lda, int arow,
    const bf16* __restrict__ Bt, int ldb, int brow, int lane)
{
  v8f acc = {0.f, 0.f, 0.f, 0.f, 0.f, 0.f, 0.f, 0.f};
  const int half = lane >> 4;
  const bf16* ap = A + arow * lda;
  const bf16* bp = Bt + brow * ldb;
#pragma unroll
  for (int k0 = 0; k0 < KPAD; k0 += 32) {
    TileU ua, ub;
    if (k0 + 16 <= KA) ua.h[0] = *(const v8bf*)(ap + k0 + half * 8);
    else { ua.w[0] = 0u; ua.w[1] = 0u; ua.w[2] = 0u; ua.w[3] = 0u; }
    if (k0 + 32 <= KA) ua.h[1] = *(const v8bf*)(ap + k0 + 16 + half * 8);
    else { ua.w[4] = 0u; ua.w[5] = 0u; ua.w[6] = 0u; ua.w[7] = 0u; }
    ub.h[0] = *(const v8bf*)(bp + k0 + half * 8);
    ub.h[1] = *(const v8bf*)(bp + k0 + 16 + half * 8);
    acc = wmma_bf16(ua.v, ub.v, acc);
  }
  return acc;
}

// ---------------------------------------------------------------------------
// Weight convert + transpose:  W[K][N] f32  ->  Wt[N][K] bf16
// ---------------------------------------------------------------------------
__global__ void tconv_kernel(const float* __restrict__ W, bf16* __restrict__ Wt, int K, int N) {
  int i = blockIdx.x * blockDim.x + threadIdx.x;
  if (i >= K * N) return;
  int n = i / K, k = i - n * K;
  Wt[i] = f2b(W[(size_t)k * N + n]);
}

// ---------------------------------------------------------------------------
// Embedding + positional encoding + word-mask.  One block per (seq, word).
// ---------------------------------------------------------------------------
__global__ __launch_bounds__(256) void embed_kernel(
    const float* __restrict__ inw, const float* __restrict__ Wemb,
    bf16* __restrict__ h, float* __restrict__ maskw)
{
  int t = blockIdx.x;        // 0 .. NS*16-1
  int c = threadIdx.x;       // 0 .. 255
  float x = inw[t];
  bool neg = (x <= 0.f);
  int idx = neg ? 1 : (int)x;
  float num = neg ? x : 0.f;
  float e = Wemb[(size_t)idx * 256 + c] - num * 0.01f;
  int w = t & 15;
  float invf = __expf(-(2.f * (float)c / 256.f) * 9.2103403719761827f); // 10000^(-2c/d)
  float ang = (float)w * invf;
  float pe = (c & 1) ? __cosf(ang) : __sinf(ang);
  h[(size_t)t * 256 + c] = f2b(e * 16.f + pe);
  if (c == 0) maskw[t] = (x != 0.f) ? 1.f : 0.f;
}

__global__ void maskq_kernel(const float* __restrict__ maskw, float* __restrict__ maskS, int ns) {
  int i = blockIdx.x * blockDim.x + threadIdx.x;
  if (i >= ns) return;
  float any = 0.f;
  for (int w = 0; w < 16; ++w) any = fmaxf(any, maskw[i * 16 + w]);
  maskS[i] = any;
}

// ---------------------------------------------------------------------------
// Fully fused word-stage transformer layer.  One block = one 16-token
// sequence; all intermediates live in LDS.  8 waves / block.
// ---------------------------------------------------------------------------
template<bool TOKENQ>
__global__ __launch_bounds__(256) void word_layer_kernel(
    const bf16* __restrict__ hin, bf16* __restrict__ hout,
    const bf16* __restrict__ Wvt, const bf16* __restrict__ Wkt,
    const bf16* __restrict__ Wqt, const bf16* __restrict__ Wot,
    const bf16* __restrict__ W1t, const bf16* __restrict__ W2t,
    const float* __restrict__ g0, const float* __restrict__ b0,
    const float* __restrict__ b1, const float* __restrict__ b2,
    const float* __restrict__ g1, const float* __restrict__ bn1,
    const float* __restrict__ maskw, const float* __restrict__ maskq,
    const float* __restrict__ token)
{
  __shared__ __align__(16) unsigned char smem[61568];
  bf16 (*sX)[256]    = (bf16 (*)[256])(smem);           // layer input (bf16)
  bf16 (*sQ)[256]    = (bf16 (*)[256])(smem + 8192);    // Q / Wo-input staging
  bf16 (*sH)[256]    = (bf16 (*)[256])(smem + 8192);    // overlays sQ
  bf16 (*sK)[256]    = (bf16 (*)[256])(smem + 16384);   // K
  bf16 (*sM)[512]    = (bf16 (*)[512])(smem + 16384);   // FFN mid, overlays sK + sVt lower half
  bf16 (*sVt)[32]    = (bf16 (*)[32])(smem + 24576);    // V^T [feature][token], keys padded to 32
  bf16 (*sP)[16][16] = (bf16 (*)[16][16])(smem + 40960);// probs [head][q][k]
  float (*sA)[256]   = (float (*)[256])(smem + 45056);  // f32 accum
  float* sMask = (float*)(smem + 61440);
  float* sMqp  = (float*)(smem + 61504);

  const int ns = blockIdx.x;
  const int tid = threadIdx.x;
  const int wave = tid >> 5, lane = tid & 31, lr = lane & 15;

  // ---- async copy x (8KB) global -> LDS (ASYNCcnt path), masks, token ----
  const bf16* xg = hin + (size_t)ns * 4096;
  {
    bf16* lp = &sX[0][0] + tid * 16;
    const bf16* gp = xg + tid * 16;
    unsigned ldsAddr = (unsigned)(unsigned long long)lp;   // low 32 bits = LDS offset
    asm volatile(
      "global_load_async_to_lds_b128 %0, %1, off\n\t"
      "global_load_async_to_lds_b128 %0, %1, off offset:16"
      :: "v"(ldsAddr), "v"(gp)
      : "memory");
  }
  if (tid < 16) sMask[tid] = maskw[ns * 16 + tid];
  // zero-pad sVt keys 16..31 (so attention P@V can run with KPAD=32 unguarded)
  for (int i = tid; i < 4096; i += 256) { int r = i >> 4, c = (i & 15) + 16; sVt[r][c] = f2b(0.f); }
  if (TOKENQ) {
    if (tid == 0) *sMqp = maskq[ns];
    for (int i = tid; i < 4096; i += 256) sQ[0][i] = f2b(0.f);
    if (tid < 256) sQ[0][tid] = f2b(token[tid]);   // row 0 = token vector
  }
  asm volatile("s_wait_asynccnt 0x0" ::: "memory");
  __syncthreads();

  // ---- V, K, (Q) projections: X(16x256) @ Wt^T ----
  for (int t = wave; t < 16; t += 8) {
    int n0 = t * 16;
    int col = n0 + lr, rb = (lane >> 4) * 8;
    {
      v8f acc = wave_gemm_tile<256, 256>(&sX[0][0], 256, lr, Wvt, 256, n0 + lr, lane);
#pragma unroll
      for (int j = 0; j < 8; ++j) sVt[col][rb + j] = f2b(acc[j]);   // transposed store
    }
    {
      v8f acc = wave_gemm_tile<256, 256>(&sX[0][0], 256, lr, Wkt, 256, n0 + lr, lane);
#pragma unroll
      for (int j = 0; j < 8; ++j) sK[rb + j][col] = f2b(acc[j]);
    }
    if (!TOKENQ) {
      v8f acc = wave_gemm_tile<256, 256>(&sX[0][0], 256, lr, Wqt, 256, n0 + lr, lane);
#pragma unroll
      for (int j = 0; j < 8; ++j) sQ[rb + j][col] = f2b(acc[j]);
    }
  }
  __syncthreads();

  // ---- energy^T = K @ Q^T per head (wave = head), masked softmax over keys ----
  {
    v8f e = wave_gemm_tile<32, 32>(&sK[0][wave * 32], 256, lr, &sQ[0][wave * 32], 256, lr, lane);
    const int qc = lr, kb = (lane >> 4) * 8;
    float ev[8];
#pragma unroll
    for (int j = 0; j < 8; ++j)
      ev[j] = (sMask[kb + j] != 0.f) ? e[j] * 0.0625f : -6.25e18f;  // /sqrt(256); -1e20/16
    float mx = ev[0];
#pragma unroll
    for (int j = 1; j < 8; ++j) mx = fmaxf(mx, ev[j]);
    mx = fmaxf(mx, __shfl_xor(mx, 16, 32));
    float p[8], s = 0.f;
#pragma unroll
    for (int j = 0; j < 8; ++j) { p[j] = __expf(ev[j] - mx); s += p[j]; }
    s += __shfl_xor(s, 16, 32);
    float inv = 1.f / s;
#pragma unroll
    for (int j = 0; j < 8; ++j) sP[wave][qc][kb + j] = f2b(p[j] * inv);
  }
  __syncthreads();

  // ---- attention output = P @ V per head (KPAD=32, A valid K=16, B padded) ----
  for (int nt = 0; nt < 2; ++nt) {
    int f0 = wave * 32 + nt * 16;
    v8f acc = wave_gemm_tile<32, 16>(&sP[wave][0][0], 16, lr, &sVt[f0][0], 32, lr, lane);
    int col = f0 + lr, rb = (lane >> 4) * 8;
#pragma unroll
    for (int j = 0; j < 8; ++j) sA[rb + j][col] = acc[j];
  }
  __syncthreads();

  // ---- convert attn out to bf16 (into sQ) for Wo GEMM ----
  for (int i = tid; i < 4096; i += 256) sQ[0][i] = f2b(sA[0][i]);
  __syncthreads();

  // ---- Wo projection -> sA (f32) ----
  for (int t = wave; t < 16; t += 8) {
    int n0 = t * 16;
    v8f acc = wave_gemm_tile<256, 256>(&sQ[0][0], 256, lr, Wot, 256, n0 + lr, lane);
    int col = n0 + lr, rb = (lane >> 4) * 8;
#pragma unroll
    for (int j = 0; j < 8; ++j) sA[rb + j][col] = acc[j];
  }
  __syncthreads();

  // ---- LN0, mask_q, h = a + q  -> sH (bf16) ----
  {
    const int row = tid >> 4, c0 = tid & 15;
    float v[16], sum = 0.f;
#pragma unroll
    for (int i = 0; i < 16; ++i) { v[i] = sA[row][c0 + 16 * i]; sum += v[i]; }
    sum += __shfl_xor(sum, 1, 16); sum += __shfl_xor(sum, 2, 16);
    sum += __shfl_xor(sum, 4, 16); sum += __shfl_xor(sum, 8, 16);
    float mu = sum * (1.f / 256.f), vs = 0.f;
#pragma unroll
    for (int i = 0; i < 16; ++i) { float d = v[i] - mu; vs += d * d; }
    vs += __shfl_xor(vs, 1, 16); vs += __shfl_xor(vs, 2, 16);
    vs += __shfl_xor(vs, 4, 16); vs += __shfl_xor(vs, 8, 16);
    float rstd = rsqrtf(vs * (1.f / 256.f) + 1e-5f);
    float mq = TOKENQ ? *sMqp : sMask[row];
#pragma unroll
    for (int i = 0; i < 16; ++i) {
      int c = c0 + 16 * i;
      float a = (g0[c] * (v[i] - mu) * rstd + b0[c]) * mq;
      float q = TOKENQ ? token[c] : b2f(sX[row][c]);
      sH[row][c] = f2b(a + q);
    }
  }
  __syncthreads();

  // ---- FFN1: sH @ W1t^T + b1, quickgelu -> sM ----
  for (int t = wave; t < 32; t += 8) {
    int n0 = t * 16;
    v8f acc = wave_gemm_tile<256, 256>(&sH[0][0], 256, lr, W1t, 256, n0 + lr, lane);
    int col = n0 + lr, rb = (lane >> 4) * 8;
    float bb = b1[col];
#pragma unroll
    for (int j = 0; j < 8; ++j) {
      float x = acc[j] + bb;
      sM[rb + j][col] = f2b(x / (1.f + __expf(-1.702f * x)));
    }
  }
  __syncthreads();

  // ---- FFN2: sM @ W2t^T + b2 -> sA (f32) ----
  for (int t = wave; t < 16; t += 8) {
    int n0 = t * 16;
    v8f acc = wave_gemm_tile<512, 512>(&sM[0][0], 512, lr, W2t, 512, n0 + lr, lane);
    int col = n0 + lr, rb = (lane >> 4) * 8;
    float bb = b2[col];
#pragma unroll
    for (int j = 0; j < 8; ++j) sA[rb + j][col] = acc[j] + bb;
  }
  __syncthreads();

  // ---- LN1 (+residual if !TOKENQ), mask, store ----
  {
    const int row = tid >> 4, c0 = tid & 15;
    float v[16], sum = 0.f;
#pragma unroll
    for (int i = 0; i < 16; ++i) {
      int c = c0 + 16 * i;
      float q = TOKENQ ? 0.f : b2f(sX[row][c]);
      v[i] = sA[row][c] + q;
      sum += v[i];
    }
    sum += __shfl_xor(sum, 1, 16); sum += __shfl_xor(sum, 2, 16);
    sum += __shfl_xor(sum, 4, 16); sum += __shfl_xor(sum, 8, 16);
    float mu = sum * (1.f / 256.f), vs = 0.f;
#pragma unroll
    for (int i = 0; i < 16; ++i) { float d = v[i] - mu; vs += d * d; }
    vs += __shfl_xor(vs, 1, 16); vs += __shfl_xor(vs, 2, 16);
    vs += __shfl_xor(vs, 4, 16); vs += __shfl_xor(vs, 8, 16);
    float rstd = rsqrtf(vs * (1.f / 256.f) + 1e-5f);
    float mq = TOKENQ ? *sMqp : sMask[row];
    if (!TOKENQ) {
#pragma unroll
      for (int i = 0; i < 16; ++i) {
        int c = c0 + 16 * i;
        hout[(size_t)ns * 4096 + row * 256 + c] = f2b((g1[c] * (v[i] - mu) * rstd + bn1[c]) * mq);
      }
    } else if (row == 0) {
#pragma unroll
      for (int i = 0; i < 16; ++i) {
        int c = c0 + 16 * i;
        hout[(size_t)ns * 256 + c] = f2b((g1[c] * (v[i] - mu) * rstd + bn1[c]) * mq);
      }
    }
  }
}

// ---------------------------------------------------------------------------
// Generic batched WMMA GEMM:  C[b] = act(A[b] @ Bt[b]^T + bias)
// K compile-time; MG = M not multiple of 16 (row clamp + guarded store).
// Grid: (ceil(M/16), ceil(N/128), B1*B2); 8 waves, one 16x16 tile each.
// ---------------------------------------------------------------------------
template<int K, bool MG, bool ACT>
__global__ __launch_bounds__(256) void gemm_bt_kernel(
    const bf16* __restrict__ A, long long a1, long long a2, int lda,
    const bf16* __restrict__ B, long long bs1, long long bs2, int ldb,
    bf16* __restrict__ C, long long c1, long long c2, int ldc,
    const float* __restrict__ bias, int M, int N, int nB2)
{
  const int wave = threadIdx.x >> 5, lane = threadIdx.x & 31, lr = lane & 15;
  const int n0 = (blockIdx.y * 8 + wave) * 16;
  if (n0 >= N) return;                         // wave-uniform
  const int z = blockIdx.z, zb1 = z / nB2, zb2 = z - zb1 * nB2;
  const bf16* Ab = A + (size_t)zb1 * a1 + (size_t)zb2 * a2;
  const bf16* Bb = B + (size_t)zb1 * bs1 + (size_t)zb2 * bs2;
  bf16* Cb = C + (size_t)zb1 * c1 + (size_t)zb2 * c2;
  const int m0 = blockIdx.x * 16;
  int arow = m0 + lr;
  if (MG && arow >= M) arow = 0;               // clamped; extra rows not stored
  const int brow = n0 + lr;
  __builtin_prefetch(Bb + (size_t)brow * ldb, 0, 1);   // global_prefetch_b8
  v8f acc = wave_gemm_tile<K, K>(Ab, lda, arow, Bb, ldb, brow, lane);
  const int col = n0 + lr;
  const int rb = m0 + (lane >> 4) * 8;
  float bb = bias ? bias[col] : 0.f;
#pragma unroll
  for (int j = 0; j < 8; ++j) {
    int r = rb + j;
    if (!MG || r < M) {
      float x = acc[j] + bb;
      if (ACT) x = x / (1.f + __expf(-1.702f * x));
      Cb[(size_t)r * ldc + col] = f2b(x);
    }
  }
}

// Row softmax over Lk with key mask; in place, one wave per row.
__global__ __launch_bounds__(256) void softmax_kernel(
    bf16* __restrict__ E, const float* __restrict__ maskk,
    int Lk, int rowsPerN, int totalRows)
{
  int row = blockIdx.x * 8 + (threadIdx.x >> 5);
  if (row >= totalRows) return;
  int lane = threadIdx.x & 31;
  int n = row / rowsPerN;
  bf16* p = E + (size_t)row * Lk;
  const float* mk = maskk + (size_t)n * Lk;
  float mx = -3.0e38f;
  for (int k = lane; k < Lk; k += 32) {
    float e = (mk[k] != 0.f) ? b2f(p[k]) * 0.0625f : -6.25e18f;
    mx = fmaxf(mx, e);
  }
  for (int o = 16; o; o >>= 1) mx = fmaxf(mx, __shfl_xor(mx, o, 32));
  float s = 0.f;
  for (int k = lane; k < Lk; k += 32) {
    float e = (mk[k] != 0.f) ? b2f(p[k]) * 0.0625f : -6.25e18f;
    s += __expf(e - mx);
  }
  for (int o = 16; o; o >>= 1) s += __shfl_xor(s, o, 32);
  float inv = 1.f / s;
  for (int k = lane; k < Lk; k += 32) {
    float e = (mk[k] != 0.f) ? b2f(p[k]) * 0.0625f : -6.25e18f;
    p[k] = f2b(__expf(e - mx) * inv);
  }
}

// Generic row LayerNorm (d=256), one wave per row:
//   xin = a (+ q if resInside); out = LN(xin)*g + b, *mask, (+ q if postAdd)
__global__ __launch_bounds__(256) void ln_kernel(
    const bf16* __restrict__ a, const bf16* __restrict__ q,
    const float* __restrict__ g, const float* __restrict__ b,
    const float* __restrict__ maskrow,
    bf16* __restrict__ outb, float* __restrict__ outf,
    int rows, int resInside, int postAdd)
{
  int row = blockIdx.x * 8 + (threadIdx.x >> 5);
  if (row >= rows) return;
  int lane = threadIdx.x & 31;
  const bf16* ar = a + (size_t)row * 256;
  const bf16* qr = q ? q + (size_t)row * 256 : nullptr;
  float v[8], sum = 0.f;
#pragma unroll
  for (int i = 0; i < 8; ++i) {
    int c = lane + 32 * i;
    float x = b2f(ar[c]);
    if (resInside && qr) x += b2f(qr[c]);
    v[i] = x; sum += x;
  }
  for (int o = 16; o; o >>= 1) sum += __shfl_xor(sum, o, 32);
  float mu = sum * (1.f / 256.f), vs = 0.f;
#pragma unroll
  for (int i = 0; i < 8; ++i) { float d = v[i] - mu; vs += d * d; }
  for (int o = 16; o; o >>= 1) vs += __shfl_xor(vs, o, 32);
  float rstd = rsqrtf(vs * (1.f / 256.f) + 1e-5f);
  float mq = maskrow ? maskrow[row] : 1.f;
#pragma unroll
  for (int i = 0; i < 8; ++i) {
    int c = lane + 32 * i;
    float ov = (g[c] * (v[i] - mu) * rstd + b[c]) * mq;
    if (postAdd && qr) ov += b2f(qr[c]);
    if (outb) outb[(size_t)row * 256 + c] = f2b(ov);
    else      outf[(size_t)row * 256 + c] = ov;
  }
}

// V[16][512][256] -> Vt[16][256][512]
__global__ __launch_bounds__(256) void vtrans_kernel(const bf16* __restrict__ V, bf16* __restrict__ Vt) {
  size_t i = (size_t)blockIdx.x * 256 + threadIdx.x;
  int s = (int)(i & 511), f = (int)((i >> 9) & 255), n = (int)(i >> 17);
  Vt[i] = V[((size_t)n * 512 + s) * 256 + f];
}

// tok[n][c] = sum_s s[n][s][c] / count_valid(n)
__global__ __launch_bounds__(256) void pool_kernel(const bf16* __restrict__ s,
                                                   const float* __restrict__ maskS,
                                                   bf16* __restrict__ tok) {
  int n = blockIdx.x, c = threadIdx.x;
  float sum = 0.f, cnt = 0.f;
  for (int i = 0; i < 512; ++i) {
    sum += b2f(s[((size_t)n * 512 + i) * 256 + c]);
    cnt += maskS[n * 512 + i];
  }
  tok[n * 256 + c] = f2b(sum / cnt);
}

// ---------------------------------------------------------------------------
// Host side
// ---------------------------------------------------------------------------
struct LP { const float *W1,*W2,*Wk,*Wo,*Wq,*Wv,*b0,*b1,*b2,*bn1,*g0,*g1; };
struct WT { bf16 *Wvt,*Wkt,*Wqt,*Wot,*W1t,*W2t; };
struct SB { bf16 *V,*K,*Q,*Vt,*E,*aO,*aW,*h1,*mid,*ff; };

static LP get_lp(void* const* d_in, int base, bool hasWq) {
  LP p; int i = base;
  p.W1 = (const float*)d_in[i++]; p.W2 = (const float*)d_in[i++];
  p.Wk = (const float*)d_in[i++]; p.Wo = (const float*)d_in[i++];
  p.Wq = hasWq ? (const float*)d_in[i++] : nullptr;
  p.Wv = (const float*)d_in[i++];
  p.b0 = (const float*)d_in[i++]; p.b1 = (const float*)d_in[i++];
  p.b2 = (const float*)d_in[i++]; p.bn1 = (const float*)d_in[i++];
  p.g0 = (const float*)d_in[i++]; p.g1 = (const float*)d_in[i++];
  return p;
}

static inline void launch_tconv(const float* W, bf16* Wt, int K, int N, hipStream_t s) {
  int total = K * N;
  tconv_kernel<<<(total + 255) / 256, 256, 0, s>>>(W, Wt, K, N);
}

static inline void launch_gemm(const bf16* A, long long a1, long long a2, int lda,
                               const bf16* B, long long b1, long long b2, int ldb,
                               bf16* C, long long c1, long long c2, int ldc,
                               const float* bias, int M, int N, int K,
                               int B1, int B2, int act, hipStream_t s) {
  dim3 g((M + 15) / 16, (N + 127) / 128, B1 * B2);
  const bool mg = (M & 15) != 0;
  if (K == 256) {
    if (act) gemm_bt_kernel<256, false, true ><<<g, 256, 0, s>>>(A,a1,a2,lda, B,b1,b2,ldb, C,c1,c2,ldc, bias, M, N, B2);
    else     gemm_bt_kernel<256, false, false><<<g, 256, 0, s>>>(A,a1,a2,lda, B,b1,b2,ldb, C,c1,c2,ldc, bias, M, N, B2);
  } else if (K == 512) {
    if (mg) gemm_bt_kernel<512, true,  false><<<g, 256, 0, s>>>(A,a1,a2,lda, B,b1,b2,ldb, C,c1,c2,ldc, bias, M, N, B2);
    else    gemm_bt_kernel<512, false, false><<<g, 256, 0, s>>>(A,a1,a2,lda, B,b1,b2,ldb, C,c1,c2,ldc, bias, M, N, B2);
  } else { // K == 32
    if (mg) gemm_bt_kernel<32, true,  false><<<g, 256, 0, s>>>(A,a1,a2,lda, B,b1,b2,ldb, C,c1,c2,ldc, bias, M, N, B2);
    else    gemm_bt_kernel<32, false, false><<<g, 256, 0, s>>>(A,a1,a2,lda, B,b1,b2,ldb, C,c1,c2,ldc, bias, M, N, B2);
  }
}

static void run_sentence_layer(const bf16* x, bf16* out, const WT& w, const LP& p,
                               const float* maskS, const SB& B, int postAdd, hipStream_t s)
{
  // projections (M = 16*512 rows flattened)
  launch_gemm(x,0,0,256, w.Wvt,0,0,256, B.V,0,0,256, nullptr, 8192,256,256, 1,1,0, s);
  launch_gemm(x,0,0,256, w.Wkt,0,0,256, B.K,0,0,256, nullptr, 8192,256,256, 1,1,0, s);
  launch_gemm(x,0,0,256, w.Wqt,0,0,256, B.Q,0,0,256, nullptr, 8192,256,256, 1,1,0, s);
  vtrans_kernel<<<8192, 256, 0, s>>>(B.V, B.Vt);
  // energy E[n,h,q,k] = Q . K  (per (n,h): M=512,N=512,K=32)
  launch_gemm(B.Q,(long long)512*256,32,256, B.K,(long long)512*256,32,256,
              B.E,(long long)8*512*512,(long long)512*512,512, nullptr,
              512,512,32, 16,8,0, s);
  softmax_kernel<<<65536/8, 256, 0, s>>>(B.E, maskS, 512, 4096, 65536);
  // out[n,q,h*32+f] = P @ V  (per (n,h): M=512,N=32,K=512)
  launch_gemm(B.E,(long long)8*512*512,(long long)512*512,512,
              B.Vt,(long long)256*512,(long long)32*512,512,
              B.aO,(long long)512*256,32,256, nullptr, 512,32,512, 16,8,0, s);
  // Wo
  launch_gemm(B.aO,0,0,256, w.Wot,0,0,256, B.aW,0,0,256, nullptr, 8192,256,256, 1,1,0, s);
  // h1 = LN(aW)*g0+b0 * mask + x
  ln_kernel<<<1024, 256, 0, s>>>(B.aW, x, p.g0, p.b0, maskS, B.h1, nullptr, 8192, 0, 1);
  // FFN
  launch_gemm(B.h1,0,0,256, w.W1t,0,0,256, B.mid,0,0,512, p.b1, 8192,512,256, 1,1,1, s);
  launch_gemm(B.mid,0,0,512, w.W2t,0,0,512, B.ff,0,0,256, p.b2, 8192,256,512, 1,1,0, s);
  // out = LN(ff + x)*g1+bn1 * mask (+ x if postAdd)
  ln_kernel<<<1024, 256, 0, s>>>(B.ff, x, p.g1, p.bn1, maskS, out, nullptr, 8192, 1, postAdd);
}

extern "C" void kernel_launch(void* const* d_in, const int* in_sizes, int n_in,
                              void* d_out, int out_size, void* d_ws, size_t ws_size,
                              hipStream_t stream)
{
  (void)in_sizes; (void)n_in; (void)out_size; (void)ws_size;
  // jax pytree (sorted-key) flatten order:
  // 0 inputwords, 1 Wemb, s0@2, s1@14, s2@26, s3@38, token_w0@50, w0@51, w1@63, w2@75, w3@87
  const float* inw   = (const float*)d_in[0];
  const float* Wemb  = (const float*)d_in[1];
  const float* token = (const float*)d_in[50];
  LP lps[8];
  lps[0] = get_lp(d_in, 51, true);   // w0
  lps[1] = get_lp(d_in, 63, true);   // w1
  lps[2] = get_lp(d_in, 75, true);   // w2
  lps[3] = get_lp(d_in, 87, false);  // w3 (no Wq)
  lps[4] = get_lp(d_in, 2,  true);   // s0
  lps[5] = get_lp(d_in, 14, true);   // s1
  lps[6] = get_lp(d_in, 26, true);   // s2
  lps[7] = get_lp(d_in, 38, true);   // s3

  char* wsb = (char*)d_ws;
  size_t off = 0;
  auto alloc = [&](size_t nbytes) -> void* {
    void* p = wsb + off;
    off += (nbytes + 255) & ~(size_t)255;
    return p;
  };

  WT wts[8];
  for (int i = 0; i < 8; ++i) {
    wts[i].Wvt = (bf16*)alloc(65536 * 2);  wts[i].Wkt = (bf16*)alloc(65536 * 2);
    wts[i].Wqt = (bf16*)alloc(65536 * 2);  wts[i].Wot = (bf16*)alloc(65536 * 2);
    wts[i].W1t = (bf16*)alloc(131072 * 2); wts[i].W2t = (bf16*)alloc(131072 * 2);
  }
  bf16* h_w    = (bf16*)alloc((size_t)8192 * 4096 * 2);
  float* maskw = (float*)alloc(131072 * 4);
  float* maskS = (float*)alloc(8192 * 4);
  bf16* h_s    = (bf16*)alloc((size_t)8192 * 256 * 2);
  bf16* h_s2   = (bf16*)alloc((size_t)8192 * 256 * 2);
  SB sb;
  sb.V   = (bf16*)alloc((size_t)2097152 * 2);
  sb.K   = (bf16*)alloc((size_t)2097152 * 2);
  sb.Q   = (bf16*)alloc((size_t)2097152 * 2);
  sb.Vt  = (bf16*)alloc((size_t)2097152 * 2);
  sb.E   = (bf16*)alloc((size_t)33554432 * 2);
  sb.aO  = (bf16*)alloc((size_t)2097152 * 2);
  sb.aW  = (bf16*)alloc((size_t)2097152 * 2);
  sb.h1  = (bf16*)alloc((size_t)2097152 * 2);
  sb.mid = (bf16*)alloc((size_t)4194304 * 2);
  sb.ff  = (bf16*)alloc((size_t)2097152 * 2);
  bf16* tokb = (bf16*)alloc(4096 * 2);
  bf16* Qp   = (bf16*)alloc(4096 * 2);
  bf16* E2   = (bf16*)alloc(65536 * 2);
  bf16* aO2  = (bf16*)alloc(4096 * 2);
  bf16* aW2  = (bf16*)alloc(4096 * 2);
  bf16* h1f  = (bf16*)alloc(4096 * 2);
  bf16* mid2 = (bf16*)alloc(8192 * 2);
  bf16* ff2  = (bf16*)alloc(4096 * 2);

  // ---- convert + transpose weights (f32 -> bf16, [K][N] -> [N][K]) ----
  for (int i = 0; i < 8; ++i) {
    launch_tconv(lps[i].Wv, wts[i].Wvt, 256, 256, stream);
    launch_tconv(lps[i].Wk, wts[i].Wkt, 256, 256, stream);
    if (lps[i].Wq) launch_tconv(lps[i].Wq, wts[i].Wqt, 256, 256, stream);
    launch_tconv(lps[i].Wo, wts[i].Wot, 256, 256, stream);
    launch_tconv(lps[i].W1, wts[i].W1t, 256, 512, stream);
    launch_tconv(lps[i].W2, wts[i].W2t, 512, 256, stream);
  }

  // ---- embedding + PE + masks ----
  embed_kernel<<<131072, 256, 0, stream>>>(inw, Wemb, h_w, maskw);
  maskq_kernel<<<32, 256, 0, stream>>>(maskw, maskS, 8192);

  // ---- word stage: w0..w2 fused (in place), then w3 (token query -> h_s) ----
  for (int l = 0; l < 3; ++l)
    word_layer_kernel<false><<<8192, 256, 0, stream>>>(h_w, h_w,
        wts[l].Wvt, wts[l].Wkt, wts[l].Wqt, wts[l].Wot, wts[l].W1t, wts[l].W2t,
        lps[l].g0, lps[l].b0, lps[l].b1, lps[l].b2, lps[l].g1, lps[l].bn1,
        maskw, nullptr, nullptr);
  word_layer_kernel<true><<<8192, 256, 0, stream>>>(h_w, h_s,
      wts[3].Wvt, wts[3].Wkt, wts[3].Wkt /*unused*/, wts[3].Wot, wts[3].W1t, wts[3].W2t,
      lps[3].g0, lps[3].b0, lps[3].b1, lps[3].b2, lps[3].g1, lps[3].bn1,
      maskw, maskS, token);

  // ---- sentence stage s0..s2 (s0 has "+ outword" post-add) ----
  run_sentence_layer(h_s,  h_s2, wts[4], lps[4], maskS, sb, 1, stream);
  run_sentence_layer(h_s2, h_s,  wts[5], lps[5], maskS, sb, 0, stream);
  run_sentence_layer(h_s,  h_s2, wts[6], lps[6], maskS, sb, 0, stream);

  // ---- masked mean pool -> tok_s ----
  pool_kernel<<<16, 256, 0, stream>>>(h_s2, maskS, tokb);

  // ---- final layer s3 (Lq=1, q = tok_s, mask_q = None) -> d_out (f32) ----
  const bf16* x = h_s2;
  const WT& w7 = wts[7]; const LP& p7 = lps[7];
  launch_gemm(x,0,0,256, w7.Wvt,0,0,256, sb.V,0,0,256, nullptr, 8192,256,256, 1,1,0, stream);
  launch_gemm(x,0,0,256, w7.Wkt,0,0,256, sb.K,0,0,256, nullptr, 8192,256,256, 1,1,0, stream);
  vtrans_kernel<<<8192, 256, 0, stream>>>(sb.V, sb.Vt);
  launch_gemm(tokb,0,0,256, w7.Wqt,0,0,256, Qp,0,0,256, nullptr, 16,256,256, 1,1,0, stream);
  // energy per (n,h): M=1, N=512, K=32
  launch_gemm(Qp,256,32,256, sb.K,(long long)512*256,32,256,
              E2,(long long)8*512,512,512, nullptr, 1,512,32, 16,8,0, stream);
  softmax_kernel<<<16, 256, 0, stream>>>(E2, maskS, 512, 8, 128);
  launch_gemm(E2,(long long)8*512,512,512, sb.Vt,(long long)256*512,(long long)32*512,512,
              aO2,256,32,256, nullptr, 1,32,512, 16,8,0, stream);
  launch_gemm(aO2,0,0,256, w7.Wot,0,0,256, aW2,0,0,256, nullptr, 16,256,256, 1,1,0, stream);
  ln_kernel<<<2, 256, 0, stream>>>(aW2, tokb, p7.g0, p7.b0, nullptr, h1f, nullptr, 16, 0, 1);
  launch_gemm(h1f,0,0,256, w7.W1t,0,0,256, mid2,0,0,512, p7.b1, 16,512,256, 1,1,1, stream);
  launch_gemm(mid2,0,0,512, w7.W2t,0,0,512, ff2,0,0,256, p7.b2, 16,256,512, 1,1,0, stream);
  ln_kernel<<<2, 256, 0, stream>>>(ff2, tokb, p7.g1, p7.bn1, nullptr, nullptr,
                                   (float*)d_out, 16, 1, 0);
}
